// HardwareReindexBranch_77481210020133
// MI455X (gfx1250) — matrix operand order repaired and checked by reading, not compiled, not run on hardware
//
#include <hip/hip_runtime.h>

typedef _Float16 v16h __attribute__((ext_vector_type(16)));
typedef float    v8f  __attribute__((ext_vector_type(8)));
typedef unsigned int u32x4 __attribute__((ext_vector_type(4)));
typedef int          i32x8 __attribute__((ext_vector_type(8)));
typedef int          i32x4 __attribute__((ext_vector_type(4)));

#define NODES 27
#define NN    (NODES*NODES)     // 729
#define HDIM  128
#define BPB   8                 // batch elements per block
#define ITERS 20
#define BIGF  3.402823466e38f

#if defined(__has_builtin)
# if __has_builtin(__builtin_amdgcn_tensor_load_to_lds) && __has_builtin(__builtin_amdgcn_s_wait_tensorcnt)
#  define USE_TDM 1
# endif
#endif
#ifndef USE_TDM
# define USE_TDM 0
#endif

__device__ __forceinline__ float wsum(float v) {
  #pragma unroll
  for (int o = 16; o; o >>= 1) v += __shfl_xor(v, o, 32);
  return v;
}
__device__ __forceinline__ float wmaxr(float v) {
  #pragma unroll
  for (int o = 16; o; o >>= 1) v = fmaxf(v, __shfl_xor(v, o, 32));
  return v;
}
__device__ __forceinline__ float wminr(float v) {
  #pragma unroll
  for (int o = 16; o; o >>= 1) v = fminf(v, __shfl_xor(v, o, 32));
  return v;
}

#if USE_TDM
// Linear 1-D tile DMA: ndwords f32 from global -> LDS (D# per ISA 08_async_tensor §8).
// 6-arg builtin form (this toolchain): (u32x4 g0, i32x8 g1, i32x4 g2, i32x4 g3, i32x8 pad, i32 cpol)
__device__ __forceinline__ void tdm_load_1d(const void* gsrc, unsigned lds_off, unsigned ndw) {
  unsigned long long ga = (unsigned long long)(uintptr_t)gsrc;
  u32x4 g0;
  g0[0] = 1u;                                            // count=1, user mode
  g0[1] = lds_off;                                       // lds_addr (bytes)
  g0[2] = (unsigned)(ga & 0xFFFFFFFFu);                  // global_addr[31:0]
  g0[3] = (unsigned)((ga >> 32) & 0x01FFFFFFu) | (2u << 30); // addr[56:32] | type=2
  i32x8 g1;
  g1[0] = (int)(2u << 16);                               // wg_mask=0, data_size=2 (4B)
  g1[1] = (int)((ndw & 0xFFFFu) << 16);                  // tensor_dim0[15:0] @ bits 63:48
  g1[2] = (int)(((ndw >> 16) & 0xFFFFu) | (1u << 16));   // tensor_dim0[31:16] | tensor_dim1=1
  g1[3] = (int)((ndw & 0xFFFFu) << 16);                  // tile_dim0 @ bits 127:112
  g1[4] = 1;                                             // tile_dim1=1, tile_dim2=0
  g1[5] = (int)ndw;                                      // tensor_dim0_stride lo32
  g1[6] = 0;                                             // stride hi16 | dim1_stride lo16
  g1[7] = 0;
  i32x4 z4 = {0, 0, 0, 0};
  i32x8 z8 = {0, 0, 0, 0, 0, 0, 0, 0};
  __builtin_amdgcn_tensor_load_to_lds(g0, g1, z4, z4, z8, 0);
}
#endif

__global__ __launch_bounds__(256) void sinkhorn_graph_kernel(
    const float* __restrict__ Bmat, const float* __restrict__ c1,
    const float* __restrict__ c2,   const float* __restrict__ Dm,
    const float* __restrict__ ln1g, const float* __restrict__ ln1b,
    const float* __restrict__ W1,   const float* __restrict__ b1,
    const float* __restrict__ W2,   const float* __restrict__ b2,
    const float* __restrict__ ln2g, const float* __restrict__ ln2b,
    const float* __restrict__ protos, const float* __restrict__ taup,
    float* __restrict__ out, int Btot)
{
  // ---- LDS (~64.9 KB) ----
  __shared__ _Float16 sW2T[HDIM*HDIM];    // 32768 B : W2 transposed f16 [col][k]
  __shared__ _Float16 sProt[NODES*HDIM];  //  6912 B
  __shared__ _Float16 sH1[32*HDIM];       //  8192 B : h1 padded to 32 rows
  __shared__ float sW1[5*HDIM];           //  2560 B
  __shared__ float sB1[HDIM], sB2[HDIM], sG2[HDIM], sBe2[HDIM]; // 2048 B
  __shared__ float sFeat[NODES*5];        //   540 B
  __shared__ float sMu[NODES], sRinv[NODES]; // 216 B
  __shared__ float sLa[NN];               //  2916 B : log_alpha 27x27
  __shared__ float sScr[3*NN];            //  8748 B : staging; aliased as f16 h2 (32x128)

  _Float16* h2h = (_Float16*)sScr;

  const int tid  = threadIdx.x;
  const int wave = tid >> 5;
  const int lane = tid & 31;
  const int hl   = lane >> 4;
  const int cl   = lane & 15;

  // ---- one-time LDS fill ----
  for (int i = tid; i < HDIM*HDIM; i += 256) {
    int k = i >> 7, c = i & 127;
    sW2T[c*HDIM + k] = (_Float16)W2[i];
  }
  for (int i = tid; i < NODES*HDIM; i += 256) sProt[i] = (_Float16)protos[i];
  for (int i = tid; i < 5*HDIM;     i += 256) sW1[i]   = W1[i];
  if (tid < HDIM) { sB1[tid]=b1[tid]; sB2[tid]=b2[tid]; sG2[tid]=ln2g[tid]; sBe2[tid]=ln2b[tid]; }
  for (int i = tid; i < 5*HDIM; i += 256) sH1[NODES*HDIM + i] = (_Float16)0.f; // rows 27..31
  float g1r[5], b1r[5];
  #pragma unroll
  for (int k = 0; k < 5; ++k) { g1r[k] = ln1g[k]; b1r[k] = ln1b[k]; }
  const float rtau = 1.0f / taup[0];
#if USE_TDM
  const unsigned scrOff = (unsigned)(uintptr_t)(void*)sScr;
#endif
  __syncthreads();

  for (int bi = 0; bi < BPB; ++bi) {
    const int b = blockIdx.x * BPB + bi;
    if (b >= Btot) break;
    const long base = (long)b * NN;

    // ---- stage Bmat/c2/D rows into LDS ----
#if USE_TDM
    if (wave == 0) {
      tdm_load_1d(Bmat + base, scrOff,            NN);
      tdm_load_1d(c2   + base, scrOff + NN*4,     NN);
      tdm_load_1d(Dm   + base, scrOff + 2*NN*4,   NN);
      __builtin_amdgcn_s_wait_tensorcnt(0);
    }
#else
    for (int i = tid; i < NN; i += 256) {
      sScr[i]        = Bmat[base + i];
      sScr[NN + i]   = c2[base + i];
      sScr[2*NN + i] = Dm[base + i];
    }
#endif
    __syncthreads();

    // ---- features + LN1 : one row per wave, lanes reduce over 27 neighbors ----
    for (int r = wave; r < NODES; r += 8) {
      const bool act = lane < NODES;
      float bm = act ? sScr[r*NODES + lane]        : 0.f;
      float cv = act ? sScr[NN + r*NODES + lane]   : 0.f;
      float dv = act ? sScr[2*NN + r*NODES + lane] : 0.f;
      const bool nb = act && (bm > 0.f);
      float deg  = wsum(bm);
      float cntf = wsum(nb ? 1.f : 0.f);
      float s2   = wsum(nb ? cv : 0.f);
      float mn   = wminr(nb ? cv : BIGF);
      float sd   = wsum(dv);
      if (lane == 0) {
        float f[5];
        f[0] = c1[(long)b*NODES + r];
        f[1] = deg;
        f[2] = s2 / fmaxf(cntf, 1.f);
        f[3] = (cntf > 0.f) ? mn : 0.f;
        f[4] = sd * (1.0f/NODES);
        float mu = 0.f;  for (int k=0;k<5;++k) mu += f[k];  mu *= 0.2f;
        float var = 0.f; for (int k=0;k<5;++k){ float d=f[k]-mu; var += d*d; } var *= 0.2f;
        float ri = rsqrtf(var + 1e-5f);
        for (int k=0;k<5;++k) sFeat[r*5+k] = (f[k]-mu)*ri*g1r[k] + b1r[k];
      }
    }
    __syncthreads();

    // ---- h1 = relu(featLN @ W1 + b1) -> f16 LDS ----
    for (int i = tid; i < NODES*HDIM; i += 256) {
      int n = i >> 7, c = i & 127;
      float acc = sB1[c];
      #pragma unroll
      for (int k = 0; k < 5; ++k) acc += sFeat[n*5+k] * sW1[k*HDIM + c];
      sH1[i] = (_Float16)fmaxf(acc, 0.f);
    }
    __syncthreads();

    // ---- h2 = h1(32x128) @ W2(128x128) + b2 : WMMA f16 ----
    {
      const int col = 16*wave + cl;
      v8f acc0 = {}; v8f acc1 = {};
      #pragma unroll
      for (int ks = 0; ks < 4; ++ks) {
        const int kb = 32*ks + 8*hl;
        const _Float16* bp = &sW2T[col*HDIM + kb];
        v16h bv;
        #pragma unroll
        for (int e = 0; e < 8; ++e) { bv[e] = bp[e]; bv[8+e] = bp[16+e]; }
        const _Float16* a0 = &sH1[cl*HDIM + kb];
        const _Float16* a1 = &sH1[(16+cl)*HDIM + kb];
        v16h av0, av1;
        #pragma unroll
        for (int e = 0; e < 8; ++e) {
          av0[e]=a0[e]; av0[8+e]=a0[16+e];
          av1[e]=a1[e]; av1[8+e]=a1[16+e];
        }
        acc0 = __builtin_amdgcn_wmma_f32_16x16x32_f16(false, av0, false, bv, (short)0, acc0, false, false);
        acc1 = __builtin_amdgcn_wmma_f32_16x16x32_f16(false, av1, false, bv, (short)0, acc1, false, false);
      }
      const float bb = sB2[col];
      #pragma unroll
      for (int r = 0; r < 8; ++r) {
        int r0 = r + 8*hl;
        h2h[r0*HDIM + col]      = (_Float16)(acc0[r] + bb);
        h2h[(16+r0)*HDIM + col] = (_Float16)(acc1[r] + bb);
      }
    }
    __syncthreads();

    // ---- LN2 stats : one row per wave, lanes hold 4 values each ----
    for (int r = wave; r < NODES; r += 8) {
      const _Float16* p = &h2h[r*HDIM + lane*4];
      float a0 = (float)p[0], a1 = (float)p[1], a2 = (float)p[2], a3 = (float)p[3];
      float s  = wsum(a0 + a1 + a2 + a3);
      float ss = wsum(a0*a0 + a1*a1 + a2*a2 + a3*a3);
      if (lane == 0) {
        float mu  = s * (1.0f/HDIM);
        float var = fmaxf(ss*(1.0f/HDIM) - mu*mu, 0.f);
        sMu[r] = mu; sRinv[r] = rsqrtf(var + 1e-5f);
      }
    }
    __syncthreads();
    // ---- LN2 apply in place + zero pad rows 27..31 ----
    for (int i = tid; i < NODES*HDIM; i += 256) {
      int n = i >> 7, c = i & 127;
      float v = (float)h2h[i];
      h2h[i] = (_Float16)((v - sMu[n]) * sRinv[n] * sG2[c] + sBe2[c]);
    }
    for (int i = tid; i < 5*HDIM; i += 256) h2h[NODES*HDIM + i] = (_Float16)0.f;
    __syncthreads();

    // ---- la[m][n] = (h2ln[n] . protos[m]) / tau : WMMA f16, waves 0..3 ----
    if (wave < 4) {
      const int tm = wave & 1, tn = wave >> 1;
      const int m = 16*tn + cl;
      v8f acc = {};
      #pragma unroll
      for (int ks = 0; ks < 4; ++ks) {
        const int kb = 32*ks + 8*hl;
        v16h bv;
        if (m < NODES) {
          const _Float16* bp = &sProt[m*HDIM + kb];
          #pragma unroll
          for (int e = 0; e < 8; ++e) { bv[e] = bp[e]; bv[8+e] = bp[16+e]; }
        } else {
          #pragma unroll
          for (int e = 0; e < 16; ++e) bv[e] = (_Float16)0.f;
        }
        const _Float16* ap = &h2h[(16*tm + cl)*HDIM + kb];
        v16h av;
        #pragma unroll
        for (int e = 0; e < 8; ++e) { av[e]=ap[e]; av[8+e]=ap[16+e]; }
        acc = __builtin_amdgcn_wmma_f32_16x16x32_f16(false, av, false, bv, (short)0, acc, false, false);
      }
      #pragma unroll
      for (int r = 0; r < 8; ++r) {
        int n = 16*tm + r + 8*hl;
        if (n < NODES && m < NODES) sLa[m*NODES + n] = acc[r] * rtau;
      }
    }
    __syncthreads();

    // ---- log-domain Sinkhorn : wave-parallel LSE via shuffle butterflies ----
    for (int it = 0; it < ITERS; ++it) {
      // rows (axis -1)
      for (int r = wave; r < NODES; r += 8) {
        float v  = (lane < NODES) ? sLa[r*NODES + lane] : -BIGF;
        float mx = wmaxr(v);
        float e  = (lane < NODES) ? __expf(v - mx) : 0.f;
        float s  = wsum(e);
        float lse = mx + __logf(s);
        if (lane < NODES) sLa[r*NODES + lane] = v - lse;
      }
      __syncthreads();
      // cols (axis -2)
      for (int cc = wave; cc < NODES; cc += 8) {
        float v  = (lane < NODES) ? sLa[lane*NODES + cc] : -BIGF;
        float mx = wmaxr(v);
        float e  = (lane < NODES) ? __expf(v - mx) : 0.f;
        float s  = wsum(e);
        float lse = mx + __logf(s);
        if (lane < NODES) sLa[lane*NODES + cc] = v - lse;
      }
      __syncthreads();
    }

    // ---- output = exp(log_alpha), coalesced ----
    for (int i = tid; i < NN; i += 256) out[base + i] = __expf(sLa[i]);
    __syncthreads();
  }
}

extern "C" void kernel_launch(void* const* d_in, const int* in_sizes, int n_in,
                              void* d_out, int out_size, void* d_ws, size_t ws_size,
                              hipStream_t stream) {
  const float* Bmat  = (const float*)d_in[0];
  const float* c1    = (const float*)d_in[1];
  const float* c2    = (const float*)d_in[2];
  const float* Dm    = (const float*)d_in[3];
  const float* ln1g  = (const float*)d_in[4];
  const float* ln1b  = (const float*)d_in[5];
  const float* W1    = (const float*)d_in[6];
  const float* b1    = (const float*)d_in[7];
  const float* W2    = (const float*)d_in[8];
  const float* b2    = (const float*)d_in[9];
  const float* ln2g  = (const float*)d_in[10];
  const float* ln2b  = (const float*)d_in[11];
  const float* prot  = (const float*)d_in[12];
  const float* taup  = (const float*)d_in[13];
  float* out = (float*)d_out;

  const int Btot = in_sizes[1] / NODES;           // c1 is (B, N)
  const int grid = (Btot + BPB - 1) / BPB;
  sinkhorn_graph_kernel<<<grid, 256, 0, stream>>>(
      Bmat, c1, c2, Dm, ln1g, ln1b, W1, b1, W2, b2, ln2g, ln2b, prot, taup, out, Btot);
}
